// BigramModel_46523085750953
// MI455X (gfx1250) — compile-verified, matrix operand order checked by reading, not run on hardware
//
#include <hip/hip_runtime.h>
#include <hip/hip_bf16.h>

typedef __bf16 bf16_t;
typedef __attribute__((ext_vector_type(16))) __bf16 v16bf;
typedef __attribute__((ext_vector_type(8)))  float  v8f;

#define WMMA_BF16(a, b, c) \
  __builtin_amdgcn_wmma_f32_16x16x32_bf16(false, (a), false, (b), (short)0, (c), false, false)

// ---------------------------------------------------------------------------
// Fragment loader: 16-bit A/B fragment, source row-major [rows, ld] bf16.
// CDNA5 16-bit A 16x32 layout: lane = half*16+row ; row M=lane%16 ;
//   lanes 0-15: K = 0..7 and 16..23 ; lanes 16-31: K = 8..15 and 24..31.
// Two contiguous 16B chunks per lane.
// ---------------------------------------------------------------------------
__device__ __forceinline__ v16bf load_frag(const bf16_t* __restrict__ p, int ld,
                                           int r, int k, int lane) {
  int half = lane >> 4, row = lane & 15;
  const bf16_t* b0 = p + (size_t)(r + row) * ld + k + half * 8;
  union { uint4 u[2]; v16bf f; } u;
  u.u[0] = *(const uint4*)(b0);
  u.u[1] = *(const uint4*)(b0 + 16);
  return u.f;
}

// ---------------------------------------------------------------------------
// Epilogue store for one 16x16 fp32 C fragment.
// C layout: lane = half*16+col ; N = col ; VGPR v -> M = v + 8*half.
// ---------------------------------------------------------------------------
__device__ __forceinline__ void store_tile(v8f acc, int m0, int n0,
                                           const float* __restrict__ bias,
                                           const float* __restrict__ residual,
                                           float* __restrict__ Cf,
                                           bf16_t* __restrict__ Cb,
                                           int M, int N, int flags, int lane) {
  int half = lane >> 4, col = lane & 15;
#pragma unroll
  for (int v = 0; v < 8; v++) {
    int m = m0 + half * 8 + v;
    int n = n0 + col;
    float val = acc[v];
    if (bias) val += bias[n];
    if (residual) val += residual[(size_t)m * N + n];
    if (flags & 1) val = fmaxf(val, 0.f);
    size_t off = (flags & 2) ? ((size_t)n * M + m) : ((size_t)m * N + n);
    if (Cb) Cb[off] = (bf16_t)val;
    else    Cf[off] = val;
  }
}

// ---------------------------------------------------------------------------
// Generic bf16 WMMA GEMM: C[M,N] = A[M,K] * Bt[N,K]^T (+bias)(+residual)(relu)
// Block: 256 threads = 8 waves (2 M x 4 N), block tile 64x128.
// Each wave: 32x32 register tile = 2x2 WMMA tiles -> 2 b128 loads per wmma.
// flags: 1 = relu, 2 = store transposed (C index [n*M + m])
// ---------------------------------------------------------------------------
__global__ __launch_bounds__(256)
void gemm_bf16_kernel(const bf16_t* __restrict__ A, const bf16_t* __restrict__ Bt,
                      const float* __restrict__ bias, const float* __restrict__ residual,
                      float* __restrict__ Cf, bf16_t* __restrict__ Cb,
                      int M, int N, int K, int flags) {
  int lane = threadIdx.x & 31, w = threadIdx.x >> 5;
  int m0 = blockIdx.y * 64  + (w >> 2) * 32;
  int n0 = blockIdx.x * 128 + (w & 3) * 32;
  v8f acc00 = {0.f, 0.f, 0.f, 0.f, 0.f, 0.f, 0.f, 0.f};
  v8f acc01 = acc00, acc10 = acc00, acc11 = acc00;
  for (int kk = 0; kk < K; kk += 32) {
    v16bf a0 = load_frag(A, K, m0,      kk, lane);
    v16bf a1 = load_frag(A, K, m0 + 16, kk, lane);
    v16bf b0 = load_frag(Bt, K, n0,      kk, lane);
    v16bf b1 = load_frag(Bt, K, n0 + 16, kk, lane);
    acc00 = WMMA_BF16(a0, b0, acc00);
    acc01 = WMMA_BF16(a0, b1, acc01);
    acc10 = WMMA_BF16(a1, b0, acc10);
    acc11 = WMMA_BF16(a1, b1, acc11);
  }
  store_tile(acc00, m0,      n0,      bias, residual, Cf, Cb, M, N, flags, lane);
  store_tile(acc01, m0,      n0 + 16, bias, residual, Cf, Cb, M, N, flags, lane);
  store_tile(acc10, m0 + 16, n0,      bias, residual, Cf, Cb, M, N, flags, lane);
  store_tile(acc11, m0 + 16, n0 + 16, bias, residual, Cf, Cb, M, N, flags, lane);
}

// ---------------------------------------------------------------------------
// Attention: per (b,h) and 16-row tile of S: S = K Q^T * scale (WMMA),
// causal softmax (fp32, LDS), O = P V (WMMA, V pre-transposed [D, B*T]).
// Block: 256 threads = 8 waves. gridDim = (T/16, B*H).
// ---------------------------------------------------------------------------
__global__ __launch_bounds__(256)
void attn_kernel(const bf16_t* __restrict__ kbuf, const bf16_t* __restrict__ qbuf,
                 const bf16_t* __restrict__ vt, bf16_t* __restrict__ aout,
                 int B, int T, int D, int H, int HS, float scale) {
  __shared__ float S[16 * 256];
  __shared__ __align__(16) bf16_t P[16 * 264];
  const int PST = 264;

  int lane = threadIdx.x & 31, w = threadIdx.x >> 5;
  int half = lane >> 4, col = lane & 15;
  int bh = blockIdx.y, b = bh / H, hh = bh % H;
  int i0 = blockIdx.x * 16;
  int BT = B * T;

  const bf16_t* kbase = kbuf + (size_t)(b * T + i0) * D + hh * HS;
  const bf16_t* qbase = qbuf + (size_t)(b * T) * D + hh * HS;

  // ---- phase 1: S[16, T] = K_tile * Q^T * scale (skip fully-masked tiles)
  int j0 = w * 32;
  if (j0 <= i0 + 15) {
    v8f acc0 = {0.f, 0.f, 0.f, 0.f, 0.f, 0.f, 0.f, 0.f};
    v8f acc1 = acc0;
    for (int kk = 0; kk < HS; kk += 32) {
      v16bf a  = load_frag(kbase, D, 0, kk, lane);
      v16bf b0 = load_frag(qbase, D, j0, kk, lane);
      v16bf b1 = load_frag(qbase, D, j0 + 16, kk, lane);
      acc0 = WMMA_BF16(a, b0, acc0);
      acc1 = WMMA_BF16(a, b1, acc1);
    }
#pragma unroll
    for (int v = 0; v < 8; v++) {
      int r = half * 8 + v;
      S[r * 256 + j0 + col]      = acc0[v] * scale;
      S[r * 256 + j0 + 16 + col] = acc1[v] * scale;
    }
  }
  __syncthreads();

  // ---- phase 2: causal softmax over rows 2w, 2w+1 (wave32 reductions)
  for (int rr = 0; rr < 2; rr++) {
    int r = w * 2 + rr, gi = i0 + r;
    float mx = -3.0e38f;
    for (int j = lane; j < 256; j += 32)
      if (j <= gi) mx = fmaxf(mx, S[r * 256 + j]);
#pragma unroll
    for (int off = 16; off; off >>= 1) mx = fmaxf(mx, __shfl_xor(mx, off));
    float sum = 0.f;
    for (int j = lane; j < 256; j += 32) {
      float e = (j <= gi) ? __expf(S[r * 256 + j] - mx) : 0.f;
      S[r * 256 + j] = e;
      sum += e;
    }
#pragma unroll
    for (int off = 16; off; off >>= 1) sum += __shfl_xor(sum, off);
    float inv = 1.f / sum;
    for (int j = lane; j < 256; j += 32)
      P[r * PST + j] = (bf16_t)(S[r * 256 + j] * inv);
  }
  __syncthreads();

  // ---- phase 3: O[16, HS] = P[16, T] * V ; waves 0..3 own 16 cols each
  if (w < 4) {
    int n0 = w * 16;
    const bf16_t* vbase = vt + (size_t)(hh * HS) * BT + (size_t)b * T;
    v8f acc = {0.f, 0.f, 0.f, 0.f, 0.f, 0.f, 0.f, 0.f};
    for (int jb = 0; jb < 256; jb += 32) {
      v16bf a = load_frag(&P[0], PST, 0, jb, lane);
      v16bf bb = load_frag(vbase, BT, n0, jb, lane);
      acc = WMMA_BF16(a, bb, acc);
    }
#pragma unroll
    for (int v = 0; v < 8; v++)
      aout[(size_t)(b * T + i0 + half * 8 + v) * D + hh * HS + n0 + col] = (bf16_t)acc[v];
  }
}

// ---------------------------------------------------------------------------
// LayerNorm over D=512, one row per block (256 threads, 2 elems/thread), bf16 out
// ---------------------------------------------------------------------------
__global__ __launch_bounds__(256)
void ln_kernel(const float* __restrict__ x, const float* __restrict__ s,
               const float* __restrict__ bb, bf16_t* __restrict__ out) {
  int row = blockIdx.x, tid = threadIdx.x;
  const float* xr = x + (size_t)row * 512;
  float v0 = xr[tid], v1 = xr[tid + 256];
  float sum = v0 + v1, sq = v0 * v0 + v1 * v1;
  int lane = tid & 31, w = tid >> 5;
#pragma unroll
  for (int off = 16; off; off >>= 1) {
    sum += __shfl_xor(sum, off);
    sq  += __shfl_xor(sq, off);
  }
  __shared__ float r0[8], r1[8];
  if (lane == 0) { r0[w] = sum; r1[w] = sq; }
  __syncthreads();
  float ts = 0.f, tq = 0.f;
#pragma unroll
  for (int i = 0; i < 8; i++) { ts += r0[i]; tq += r1[i]; }
  float mean = ts * (1.f / 512.f);
  float var  = tq * (1.f / 512.f) - mean * mean;
  float rstd = rsqrtf(var + 1e-5f);
  bf16_t* o = out + (size_t)row * 512;
  o[tid]       = (bf16_t)((v0 - mean) * rstd * s[tid]       + bb[tid]);
  o[tid + 256] = (bf16_t)((v1 - mean) * rstd * s[tid + 256] + bb[tid + 256]);
}

// ---------------------------------------------------------------------------
// Embedding: x[row] = tok_emb[idx[row]] + pos_emb[row % T]
// ---------------------------------------------------------------------------
__global__ __launch_bounds__(256)
void embed_kernel(const int* __restrict__ idx, const float* __restrict__ tok,
                  const float* __restrict__ pos, float* __restrict__ x, int T) {
  int row = blockIdx.x, tid = threadIdx.x;
  int token = idx[row];
  int t = row % T;
  float* xr = x + (size_t)row * 512;
  const float* te = tok + (size_t)token * 512;
  const float* pe = pos + (size_t)t * 512;
  xr[tid]       = te[tid]       + pe[tid];
  xr[tid + 256] = te[tid + 256] + pe[tid + 256];
}

// ---------------------------------------------------------------------------
// Transpose + cast fp32 [K,N] -> bf16 [N,K], batched via blockIdx.z (stride K*N)
// Block (32,8); 32x32 LDS tile, coalesced both sides.
// ---------------------------------------------------------------------------
__global__ __launch_bounds__(256)
void transpose_cast_kernel(const float* __restrict__ W, bf16_t* __restrict__ Wt,
                           int K, int N) {
  __shared__ float tile[32][33];
  size_t off = (size_t)blockIdx.z * K * N;
  const float* Wz = W + off;
  bf16_t* Wtz = Wt + off;
  int n0 = blockIdx.x * 32, k0 = blockIdx.y * 32;
  for (int i = threadIdx.y; i < 32; i += 8)
    tile[i][threadIdx.x] = Wz[(size_t)(k0 + i) * N + n0 + threadIdx.x];
  __syncthreads();
  for (int i = threadIdx.y; i < 32; i += 8)
    Wtz[(size_t)(n0 + i) * K + k0 + threadIdx.x] = (bf16_t)tile[threadIdx.x][i];
}

// ---------------------------------------------------------------------------
// Per-row cross-entropy: rowloss[row] = logsumexp(logits[row]) - logits[row][tgt]
// ---------------------------------------------------------------------------
__global__ __launch_bounds__(256)
void rowloss_kernel(const float* __restrict__ logits, const int* __restrict__ tgt,
                    float* __restrict__ rowloss, int V) {
  int row = blockIdx.x, tid = threadIdx.x;
  const float* lr = logits + (size_t)row * V;
  int lane = tid & 31, w = tid >> 5;
  __shared__ float red[8];

  float mx = -3.0e38f;
  for (int j = tid; j < V; j += 256) mx = fmaxf(mx, lr[j]);
#pragma unroll
  for (int off = 16; off; off >>= 1) mx = fmaxf(mx, __shfl_xor(mx, off));
  if (lane == 0) red[w] = mx;
  __syncthreads();
  float m = -3.0e38f;
#pragma unroll
  for (int i = 0; i < 8; i++) m = fmaxf(m, red[i]);
  __syncthreads();

  float sum = 0.f;
  for (int j = tid; j < V; j += 256) sum += __expf(lr[j] - m);
#pragma unroll
  for (int off = 16; off; off >>= 1) sum += __shfl_xor(sum, off);
  if (lane == 0) red[w] = sum;
  __syncthreads();
  float ts = 0.f;
#pragma unroll
  for (int i = 0; i < 8; i++) ts += red[i];

  if (tid == 0) rowloss[row] = m + __logf(ts) - lr[tgt[row]];
}

__global__ __launch_bounds__(256)
void loss_reduce_kernel(const float* __restrict__ rowloss, float* __restrict__ loss, int n) {
  int tid = threadIdx.x, lane = tid & 31, w = tid >> 5;
  float s = 0.f;
  for (int i = tid; i < n; i += 256) s += rowloss[i];
#pragma unroll
  for (int off = 16; off; off >>= 1) s += __shfl_xor(s, off);
  __shared__ float red[8];
  if (lane == 0) red[w] = s;
  __syncthreads();
  if (tid == 0) {
    float t = 0.f;
#pragma unroll
    for (int i = 0; i < 8; i++) t += red[i];
    loss[0] = t / (float)n;
  }
}

// ---------------------------------------------------------------------------
// Host orchestration
// ---------------------------------------------------------------------------
extern "C" void kernel_launch(void* const* d_in, const int* in_sizes, int n_in,
                              void* d_out, int out_size, void* d_ws, size_t ws_size,
                              hipStream_t stream) {
  (void)in_sizes; (void)n_in; (void)out_size; (void)ws_size;
  const int Bb = 16, Tt = 256, Dd = 512, Hh = 8, Ll = 8, HSn = 64, Vv = 32000;
  const int BT = Bb * Tt, D4 = 4 * Dd;

  const int*   idx     = (const int*)  d_in[0];
  const int*   targets = (const int*)  d_in[1];
  const float* tok_emb = (const float*)d_in[2];
  const float* pos_emb = (const float*)d_in[3];
  const float* ln1_s   = (const float*)d_in[4];
  const float* ln1_b   = (const float*)d_in[5];
  const float* wk      = (const float*)d_in[6];
  const float* wq      = (const float*)d_in[7];
  const float* wv      = (const float*)d_in[8];
  const float* w_proj  = (const float*)d_in[9];
  const float* b_proj  = (const float*)d_in[10];
  const float* ln2_s   = (const float*)d_in[11];
  const float* ln2_b   = (const float*)d_in[12];
  const float* w1      = (const float*)d_in[13];
  const float* b1      = (const float*)d_in[14];
  const float* w2      = (const float*)d_in[15];
  const float* b2      = (const float*)d_in[16];
  const float* lnf_s   = (const float*)d_in[17];
  const float* lnf_b   = (const float*)d_in[18];
  const float* w_lm    = (const float*)d_in[19];
  const float* b_lm    = (const float*)d_in[20];
  float* out = (float*)d_out;                       // [BT*V] logits, then [1] loss

  // ---- bump-allocate workspace (~132 MB)
  char* wsp = (char*)d_ws;
  auto alloc = [&](size_t bytes) -> char* {
    char* p = wsp;
    wsp += (bytes + 255) & ~(size_t)255;
    return p;
  };
  bf16_t* wkt  = (bf16_t*)alloc((size_t)Ll * Dd * Dd * 2);
  bf16_t* wqt  = (bf16_t*)alloc((size_t)Ll * Dd * Dd * 2);
  bf16_t* wvt  = (bf16_t*)alloc((size_t)Ll * Dd * Dd * 2);
  bf16_t* wpt  = (bf16_t*)alloc((size_t)Ll * Dd * Dd * 2);
  bf16_t* w1t  = (bf16_t*)alloc((size_t)Ll * Dd * D4 * 2);
  bf16_t* w2t  = (bf16_t*)alloc((size_t)Ll * D4 * Dd * 2);
  bf16_t* wlmt = (bf16_t*)alloc((size_t)Vv * Dd * 2);
  float*  x    = (float*) alloc((size_t)BT * Dd * 4);
  bf16_t* hbuf = (bf16_t*)alloc((size_t)BT * Dd * 2);
  bf16_t* kbuf = (bf16_t*)alloc((size_t)BT * Dd * 2);
  bf16_t* qbuf = (bf16_t*)alloc((size_t)BT * Dd * 2);
  bf16_t* vtb  = (bf16_t*)alloc((size_t)Dd * BT * 2);
  bf16_t* aout = (bf16_t*)alloc((size_t)BT * Dd * 2);
  bf16_t* h2   = (bf16_t*)alloc((size_t)BT * Dd * 2);
  bf16_t* y1   = (bf16_t*)alloc((size_t)BT * D4 * 2);
  bf16_t* xf   = (bf16_t*)alloc((size_t)BT * Dd * 2);
  float*  rls  = (float*) alloc((size_t)BT * 4);

  dim3 tb(32, 8);
  // ---- weight transpose+cast to bf16 [N,K]
  for (int l = 0; l < Ll; l++) {
    size_t wo = (size_t)l * Hh * Dd * HSn;   // == l*D*D
    transpose_cast_kernel<<<dim3(HSn / 32, Dd / 32, Hh), tb, 0, stream>>>(wk + wo, wkt + wo, Dd, HSn);
    transpose_cast_kernel<<<dim3(HSn / 32, Dd / 32, Hh), tb, 0, stream>>>(wq + wo, wqt + wo, Dd, HSn);
    transpose_cast_kernel<<<dim3(HSn / 32, Dd / 32, Hh), tb, 0, stream>>>(wv + wo, wvt + wo, Dd, HSn);
    transpose_cast_kernel<<<dim3(Dd / 32, Dd / 32, 1), tb, 0, stream>>>(
        w_proj + (size_t)l * Dd * Dd, wpt + (size_t)l * Dd * Dd, Dd, Dd);
    transpose_cast_kernel<<<dim3(D4 / 32, Dd / 32, 1), tb, 0, stream>>>(
        w1 + (size_t)l * Dd * D4, w1t + (size_t)l * Dd * D4, Dd, D4);
    transpose_cast_kernel<<<dim3(Dd / 32, D4 / 32, 1), tb, 0, stream>>>(
        w2 + (size_t)l * D4 * Dd, w2t + (size_t)l * D4 * Dd, D4, Dd);
  }
  transpose_cast_kernel<<<dim3(Vv / 32, Dd / 32, 1), tb, 0, stream>>>(w_lm, wlmt, Dd, Vv);

  // ---- embeddings
  embed_kernel<<<BT, 256, 0, stream>>>(idx, tok_emb, pos_emb, x, Tt);

  const float scale = 0.044194173824159216f;   // 512^-0.5 (reference scales by D)
  dim3 gD(Dd / 128, BT / 64), gD4(D4 / 128, BT / 64), gV(Vv / 128, BT / 64);

  for (int l = 0; l < Ll; l++) {
    size_t wDD = (size_t)l * Dd * Dd, wD4 = (size_t)l * Dd * D4;
    // LN1 -> h (bf16)
    ln_kernel<<<BT, 256, 0, stream>>>(x, ln1_s + l * Dd, ln1_b + l * Dd, hbuf);
    // k, q, v^T
    gemm_bf16_kernel<<<gD, 256, 0, stream>>>(hbuf, wkt + wDD, nullptr, nullptr,
                                             nullptr, kbuf, BT, Dd, Dd, 0);
    gemm_bf16_kernel<<<gD, 256, 0, stream>>>(hbuf, wqt + wDD, nullptr, nullptr,
                                             nullptr, qbuf, BT, Dd, Dd, 0);
    gemm_bf16_kernel<<<gD, 256, 0, stream>>>(hbuf, wvt + wDD, nullptr, nullptr,
                                             nullptr, vtb, BT, Dd, Dd, 2 /*transposed*/);
    // attention
    attn_kernel<<<dim3(Tt / 16, Bb * Hh), 256, 0, stream>>>(kbuf, qbuf, vtb, aout,
                                                            Bb, Tt, Dd, Hh, HSn, scale);
    // x = x + attn_out @ w_proj + b_proj  (fp32, in-place residual)
    gemm_bf16_kernel<<<gD, 256, 0, stream>>>(aout, wpt + wDD, b_proj + l * Dd, x,
                                             x, nullptr, BT, Dd, Dd, 0);
    // LN2 -> h2, FFN
    ln_kernel<<<BT, 256, 0, stream>>>(x, ln2_s + l * Dd, ln2_b + l * Dd, h2);
    gemm_bf16_kernel<<<gD4, 256, 0, stream>>>(h2, w1t + wD4, b1 + l * D4, nullptr,
                                              nullptr, y1, BT, D4, Dd, 1 /*relu*/);
    gemm_bf16_kernel<<<gD, 256, 0, stream>>>(y1, w2t + wD4, b2 + l * Dd, x,
                                             x, nullptr, BT, Dd, D4, 0);
  }

  // ---- final LN + LM head + loss
  ln_kernel<<<BT, 256, 0, stream>>>(x, lnf_s, lnf_b, xf);
  gemm_bf16_kernel<<<gV, 256, 0, stream>>>(xf, wlmt, b_lm, nullptr,
                                           out, nullptr, BT, Vv, Dd, 0);
  rowloss_kernel<<<BT, 256, 0, stream>>>(out, targets, rls, Vv);
  loss_reduce_kernel<<<1, 256, 0, stream>>>(rls, out + (size_t)BT * Vv, BT);
}
